// OutLookAttn_22677427323420
// MI455X (gfx1250) — matrix operand order repaired and checked by reading, not compiled, not run on hardware
//
#include <hip/hip_runtime.h>

// ---------------- CDNA5 WMMA / TDM types ----------------
typedef __attribute__((ext_vector_type(16))) __bf16 v16bf;
typedef __attribute__((ext_vector_type(8)))  float  v8f;
typedef __attribute__((ext_vector_type(4)))  unsigned int v4u;
typedef __attribute__((ext_vector_type(8)))  int v8i;
typedef __attribute__((ext_vector_type(4)))  int v4i;

#define NTOK   200704   // 64 * 56 * 56
#define DIM    384
#define ROWSA  50176    // 64 * 784
#define NAPAD  1024     // padded attn-logit width (972 valid)
#define NHEAD  12
#define LWIN   784

__device__ __forceinline__ unsigned short f2bf(float f) {
  unsigned int u = __float_as_uint(f);
  u += 0x7FFFu + ((u >> 16) & 1u);          // round-to-nearest-even
  return (unsigned short)(u >> 16);
}
__device__ __forceinline__ float bf2f(unsigned short u) {
  return __uint_as_float((unsigned int)u << 16);
}

union FragA { uint4 q[2]; v16bf v; };
struct Frags { FragA a, b0, b1, b2, b3; };

// ---- convert 4 consecutive f32 -> 4 bf16 (vectorized) ----
__global__ void k_cvt4(const float* __restrict__ src, unsigned short* __restrict__ dst, int n4) {
  int i = blockIdx.x * blockDim.x + threadIdx.x;
  if (i >= n4) return;
  float4 f = ((const float4*)src)[i];
  ushort4 o;
  o.x = f2bf(f.x); o.y = f2bf(f.y); o.z = f2bf(f.z); o.w = f2bf(f.w);
  ((ushort4*)dst)[i] = o;
}

// ---- 2x2 average pool: x [B,56,56,384] -> xp [B*784, 384] bf16 ----
__global__ void k_pool(const float* __restrict__ x, unsigned short* __restrict__ xp) {
  int idx = blockIdx.x * blockDim.x + threadIdx.x;
  if (idx >= ROWSA * DIM) return;
  int ch  = idx % DIM;
  int row = idx / DIM;
  int b = row / LWIN, l = row % LWIN;
  int oh = l / 28, ow = l % 28;
  size_t p00 = ((size_t)b * 3136 + (size_t)(2 * oh) * 56 + (size_t)(2 * ow)) * DIM + ch;
  float s = x[p00] + x[p00 + DIM] + x[p00 + 56 * DIM] + x[p00 + 57 * DIM];
  xp[idx] = f2bf(0.25f * s);
}

// ---- pack weight [nsrc, 384] f32 -> [npad, 384] bf16 (zero pad) ----
__global__ void k_packw(const float* __restrict__ src, unsigned short* __restrict__ dst,
                        int nsrc, int npad) {
  int idx = blockIdx.x * blockDim.x + threadIdx.x;
  if (idx >= npad * DIM) return;
  int n = idx / DIM, k = idx % DIM;
  float v = (n < nsrc) ? src[n * DIM + k] : 0.f;
  dst[idx] = f2bf(v);
}

__global__ void k_packbias(const float* __restrict__ src, float* __restrict__ dst) {
  int i = blockIdx.x * blockDim.x + threadIdx.x;
  if (i < NAPAD) dst[i] = (i < NHEAD * 81) ? src[i] : 0.f;
}

// ---- load one k-step of fragments: A from global, B tile from LDS ----
__device__ __forceinline__ void load_frags(Frags& f, const unsigned short* __restrict__ arow,
                                           const unsigned short* sbrow, int k0, int half) {
  // A k order used here (contiguous, same as B): half 0 -> K 0..15; half 1 -> K 16..31
  f.a.q[0] = *(const uint4*)(arow + k0 + half * 16);
  f.a.q[1] = *(const uint4*)(arow + k0 + half * 16 + 8);
  // B k order used here (same as A): half 0 -> K 0..15; half 1 -> K 16..31 (contiguous in each W row)
  const unsigned short* bk = sbrow + k0;
  f.b0.q[0] = *(const uint4*)(bk);            f.b0.q[1] = *(const uint4*)(bk + 8);
  f.b1.q[0] = *(const uint4*)(bk + 16 * DIM); f.b1.q[1] = *(const uint4*)(bk + 16 * DIM + 8);
  f.b2.q[0] = *(const uint4*)(bk + 32 * DIM); f.b2.q[1] = *(const uint4*)(bk + 32 * DIM + 8);
  f.b3.q[0] = *(const uint4*)(bk + 48 * DIM); f.b3.q[1] = *(const uint4*)(bk + 48 * DIM + 8);
}

// ---- bf16 WMMA GEMM: C[M,N] = A[M,384] * W[N,384]^T (+bias) ----
// Block = 8 waves = 128 rows x 64 cols.  grid.x = M/128, grid.y = N/64.
// The 64x384 bf16 B tile is staged into LDS once per block by the Tensor Data
// Mover (TDM), then all waves stream A from global with a register ping-pong.
__global__ void k_gemm(const unsigned short* __restrict__ A,
                       const unsigned short* __restrict__ Bw,
                       const float* __restrict__ bias,
                       float* __restrict__ Cf,
                       unsigned short* __restrict__ Cb,
                       int ldc) {
  __shared__ __align__(16) unsigned short sB[64 * DIM];   // 48 KB

  const int lane = threadIdx.x & 31;
  const int wave = threadIdx.x >> 5;
  const int lr   = lane & 15;
  const int half = lane >> 4;
  const size_t m0 = ((size_t)blockIdx.x * 8 + wave) * 16;
  const int    n0 = blockIdx.y * 64;

  // ---- TDM: DMA the 64x384 bf16 weight tile (rows n0..n0+63) into LDS ----
  if (threadIdx.x == 0) {
    unsigned long long ga = (unsigned long long)(const void*)(Bw + (size_t)n0 * DIM);
    unsigned lds_off = (unsigned)(unsigned long long)(const void*)sB;  // LDS aperture: low 32b = LDS offset
    v4u g0;
    g0[0] = 1u;                                          // count = 1 (valid descriptor)
    g0[1] = lds_off;                                     // lds_addr
    g0[2] = (unsigned)(ga & 0xFFFFFFFFu);                // global_addr[31:0]
    g0[3] = (unsigned)((ga >> 32) & 0x01FFFFFFu) | (2u << 30); // global_addr[56:32] | type=2
    v8i g1;
    g1[0] = (int)(1u << 16);                             // data_size = 1 -> 2 bytes/elt
    g1[1] = (int)((DIM & 0xFFFFu) << 16);                // tensor_dim0[15:0] = 384
    g1[2] = (int)((64u & 0xFFFFu) << 16);                // dim0 hi=0 | tensor_dim1 lo = 64
    g1[3] = (int)((DIM & 0xFFFFu) << 16);                // dim1 hi=0 | tile_dim0 = 384
    g1[4] = (int)(64u);                                  // tile_dim1 = 64, tile_dim2 = 0
    g1[5] = (int)DIM;                                    // tensor_dim0_stride = 384
    g1[6] = 0;                                           // stride0 hi | stride1 lo
    g1[7] = 0;
    v4i z4 = {0, 0, 0, 0};
    v8i z8 = {0, 0, 0, 0, 0, 0, 0, 0};
    __builtin_amdgcn_tensor_load_to_lds(g0, g1, z4, z4, z8, 0);
    __builtin_amdgcn_s_wait_tensorcnt(0);
  }
  __syncthreads();

  const unsigned short* arow  = A + (m0 + lr) * DIM;
  const unsigned short* sbrow = sB + (size_t)lr * DIM + half * 16;

  v8f acc0 = {}, acc1 = {}, acc2 = {}, acc3 = {};

  Frags fA, fB;
  load_frags(fA, arow, sbrow, 0, half);
#pragma unroll
  for (int it = 0; it < 12; ++it) {
    Frags& cur = (it & 1) ? fB : fA;
    Frags& nxt = (it & 1) ? fA : fB;
    if (it + 1 < 12) load_frags(nxt, arow, sbrow, (it + 1) * 32, half);
    acc0 = __builtin_amdgcn_wmma_f32_16x16x32_bf16(false, cur.a.v, false, cur.b0.v, (short)0, acc0, false, false);
    acc1 = __builtin_amdgcn_wmma_f32_16x16x32_bf16(false, cur.a.v, false, cur.b1.v, (short)0, acc1, false, false);
    acc2 = __builtin_amdgcn_wmma_f32_16x16x32_bf16(false, cur.a.v, false, cur.b2.v, (short)0, acc2, false, false);
    acc3 = __builtin_amdgcn_wmma_f32_16x16x32_bf16(false, cur.a.v, false, cur.b3.v, (short)0, acc3, false, false);
  }

  // C/D layout: VGPR r holds row (half*8 + r), col = lane%16
#define STORE_TILE(ACC, J)                                                    \
  {                                                                           \
    int col  = n0 + (J) * 16 + lr;                                            \
    float bv = bias ? bias[col] : 0.f;                                        \
    _Pragma("unroll") for (int r = 0; r < 8; ++r) {                           \
      size_t off = (m0 + (size_t)(half * 8 + r)) * (size_t)ldc + col;         \
      float val  = ACC[r] + bv;                                               \
      if (Cb) Cb[off] = f2bf(val); else Cf[off] = val;                        \
    }                                                                         \
  }
  STORE_TILE(acc0, 0)
  STORE_TILE(acc1, 1)
  STORE_TILE(acc2, 2)
  STORE_TILE(acc3, 3)
#undef STORE_TILE
}

// ---- softmax over groups of 9 within attn logits [ROWSA, 1024] ----
__global__ void k_softmax(float* __restrict__ a) {
  int idx = blockIdx.x * blockDim.x + threadIdx.x;
  if (idx >= ROWSA * NHEAD * 9) return;
  int row = idx / (NHEAD * 9);
  int rem = idx % (NHEAD * 9);              // = head*9 + i
  size_t base = (size_t)row * NAPAD + (size_t)rem * 9;
  const float SCALE = 0.17677669529663687f; // (384/12)^-0.5
  float v[9];
  float m = -1e30f;
#pragma unroll
  for (int j = 0; j < 9; ++j) { v[j] = a[base + j] * SCALE; m = fmaxf(m, v[j]); }
  float s = 0.f;
#pragma unroll
  for (int j = 0; j < 9; ++j) { v[j] = __expf(v[j] - m); s += v[j]; }
  float inv = 1.f / s;
#pragma unroll
  for (int j = 0; j < 9; ++j) a[base + j] = v[j] * inv;
}

// ---- per-window o = attn(9x9) @ vp(9x32/head), fold (overlap-add) via atomics ----
__global__ void k_fold(const float* __restrict__ attn, const unsigned short* __restrict__ v,
                       float* __restrict__ ypre) {
  __shared__ float s_a[NHEAD * 81];   // 972 floats
  __shared__ float s_v[9 * DIM];      // gathered window of v, f32
  int w = blockIdx.x;
  int b  = w / LWIN, l = w % LWIN;
  int wi = l / 28,   wj = l % 28;
  int t = threadIdx.x;

  for (int i = t; i < NHEAD * 81; i += 256) s_a[i] = attn[(size_t)w * NAPAD + i];
  for (int i = t; i < 9 * DIM; i += 256) {
    int jp = i / DIM, ch = i % DIM;
    int ph = 2 * wi - 1 + jp / 3, pw = 2 * wj - 1 + jp % 3;
    float val = 0.f;
    if (ph >= 0 && ph < 56 && pw >= 0 && pw < 56)
      val = bf2f(v[((size_t)b * 3136 + (size_t)ph * 56 + pw) * DIM + ch]);
    s_v[i] = val;
  }
  __syncthreads();

  for (int i = t; i < 9 * DIM; i += 256) {
    int ij = i / DIM, ch = i % DIM;
    int head = ch >> 5;
    const float* ar = s_a + head * 81 + ij * 9;
    const float* vc = s_v + ch;
    float o = 0.f;
#pragma unroll
    for (int jp = 0; jp < 9; ++jp) o += ar[jp] * vc[jp * DIM];
    int th = 2 * wi - 1 + ij / 3, tw = 2 * wj - 1 + ij % 3;
    if (th >= 0 && th < 56 && tw >= 0 && tw < 56)
      atomicAdd(&ypre[((size_t)b * 3136 + (size_t)th * 56 + tw) * DIM + ch], o);
  }
}

extern "C" void kernel_launch(void* const* d_in, const int* in_sizes, int n_in,
                              void* d_out, int out_size, void* d_ws, size_t ws_size,
                              hipStream_t stream) {
  (void)in_sizes; (void)n_in; (void)out_size; (void)ws_size;
  const float* x  = (const float*)d_in[0];  // [64, 3136, 384]
  const float* Wv = (const float*)d_in[1];  // [384, 384]
  const float* Wa = (const float*)d_in[2];  // [972, 384]
  const float* ba = (const float*)d_in[3];  // [972]
  const float* Wp = (const float*)d_in[4];  // [384, 384]
  const float* bp = (const float*)d_in[5];  // [384]
  float* out = (float*)d_out;               // [64, 3136, 384] f32

  char* p = (char*)d_ws;
  auto take = [&](size_t bytes) { char* r = p; p += (bytes + 255) & ~(size_t)255; return r; };
  unsigned short* xb   = (unsigned short*)take((size_t)NTOK * DIM * 2);  // x bf16
  unsigned short* vb   = (unsigned short*)take((size_t)NTOK * DIM * 2);  // v bf16
  unsigned short* yb   = (unsigned short*)take((size_t)NTOK * DIM * 2);  // folded y bf16
  unsigned short* xpb  = (unsigned short*)take((size_t)ROWSA * DIM * 2); // pooled x bf16
  unsigned short* wvb  = (unsigned short*)take((size_t)DIM * DIM * 2);
  unsigned short* wpb  = (unsigned short*)take((size_t)DIM * DIM * 2);
  unsigned short* wab  = (unsigned short*)take((size_t)NAPAD * DIM * 2);
  float*          bap  = (float*)take((size_t)NAPAD * 4);
  float*          af   = (float*)take((size_t)ROWSA * NAPAD * 4);        // attn logits/probs
  float*          ypre = (float*)take((size_t)NTOK * DIM * 4);           // fold canvas f32

  // pack inputs to bf16
  k_cvt4<<<NTOK * DIM / 4 / 256, 256, 0, stream>>>(x, xb, NTOK * DIM / 4);
  k_pool<<<ROWSA * DIM / 256, 256, 0, stream>>>(x, xpb);
  k_packw<<<DIM * DIM / 256, 256, 0, stream>>>(Wv, wvb, DIM, DIM);
  k_packw<<<DIM * DIM / 256, 256, 0, stream>>>(Wp, wpb, DIM, DIM);
  k_packw<<<NAPAD * DIM / 256, 256, 0, stream>>>(Wa, wab, NHEAD * 81, NAPAD);
  k_packbias<<<NAPAD / 256, 256, 0, stream>>>(ba, bap);

  // v = x @ Wv^T  (bf16 out)
  k_gemm<<<dim3(NTOK / 128, DIM / 64), 256, 0, stream>>>(xb, wvb, nullptr, nullptr, vb, DIM);
  // a = xp @ Wa^T + ba  (f32 out, padded width 1024)
  k_gemm<<<dim3(ROWSA / 128, NAPAD / 64), 256, 0, stream>>>(xpb, wab, bap, af, nullptr, NAPAD);
  // softmax over last K^2 axis
  k_softmax<<<ROWSA * NHEAD * 9 / 256, 256, 0, stream>>>(af);
  // fold canvas = 0, then per-window attention * v with overlap-add
  (void)hipMemsetAsync(ypre, 0, (size_t)NTOK * DIM * 4, stream);
  k_fold<<<64 * LWIN, 256, 0, stream>>>(af, vb, ypre);
  // y -> bf16, out = y @ Wp^T + bp  (f32 out)
  k_cvt4<<<NTOK * DIM / 4 / 256, 256, 0, stream>>>(ypre, yb, NTOK * DIM / 4);
  k_gemm<<<dim3(NTOK / 128, DIM / 64), 256, 0, stream>>>(yb, wpb, bp, out, nullptr, DIM);
}
